// SingleViewGAT_30863634989501
// MI455X (gfx1250) — matrix operand order repaired and checked by reading, not compiled, not run on hardware
//
#include <hip/hip_runtime.h>

typedef __bf16 bf16_t;
typedef __attribute__((ext_vector_type(16))) __bf16 v16bf;
typedef __attribute__((ext_vector_type(8)))  __bf16 v8bf;
typedef __attribute__((ext_vector_type(4)))  __bf16 v4bf;
typedef __attribute__((ext_vector_type(8)))  float   v8f;

#define LEAKY 0.2f
#define BNEPS 1e-5f

// ---- CDNA5 async global->LDS staging (guarded; fallback = sync staging) ----
#if defined(__has_builtin)
#if __has_builtin(__builtin_amdgcn_global_load_async_to_lds_b128) && \
    __has_builtin(__builtin_amdgcn_s_wait_asynccnt)
#define HAVE_ASYNC_LDS 1
#endif
#endif

#ifdef HAVE_ASYNC_LDS
typedef int v4i_t __attribute__((vector_size(16)));
#define GLOBAL_AS __attribute__((address_space(1)))
#define LDS_AS    __attribute__((address_space(3)))
__device__ __forceinline__ void async_copy_b128(const void* g, void* l) {
  // Param types per compiler diagnostic: (v4i AS1*, v4i AS3*, Imm i32 offset, Imm i32 cpol).
  // AS1 ptr from full 64-bit generic address; AS3 ptr from low 32 bits of the
  // generic LDS address (== wave-relative LDS byte offset per ISA truncation).
  __builtin_amdgcn_global_load_async_to_lds_b128(
      (GLOBAL_AS v4i_t*)(unsigned long long)(size_t)g,
      (LDS_AS v4i_t*)(unsigned)(size_t)l, 0, 0);
}
#endif

// ---------------- helpers ----------------
__device__ __forceinline__ unsigned f2ord(float f) {
  unsigned u = __float_as_uint(f);
  return (u >> 31) ? ~u : (u | 0x80000000u);
}
__device__ __forceinline__ float ord2f(unsigned o) {
  return __uint_as_float((o >> 31) ? (o & 0x7fffffffu) : ~o);
}
__device__ __forceinline__ float leaky(float m) { return m > 0.f ? m : LEAKY * m; }

// ---------------- utility kernels ----------------
__global__ void fill_u32_kernel(unsigned* __restrict__ p, unsigned v, long n) {
  long i = (long)blockIdx.x * blockDim.x + threadIdx.x;
  if (i < n) p[i] = v;
}

// n must be a multiple of 4 (all our tensors are)
__global__ void f32_to_bf16_kernel(const float* __restrict__ in,
                                   bf16_t* __restrict__ out, long n) {
  long i = ((long)blockIdx.x * blockDim.x + threadIdx.x) * 4;
  if (i >= n) return;
  float4 v = *(const float4*)(in + i);
  v4bf o;
  o.x = (bf16_t)v.x; o.y = (bf16_t)v.y; o.z = (bf16_t)v.z; o.w = (bf16_t)v.w;
  *(v4bf*)(out + i) = o;
}

// convert + transpose: in is R x C row-major fp32, out is C x R row-major bf16
__global__ void f32_to_bf16_T_kernel(const float* __restrict__ in, bf16_t* __restrict__ out,
                                     int R, int Ccols) {
  long i = (long)blockIdx.x * blockDim.x + threadIdx.x;
  if (i >= (long)R * Ccols) return;
  int r = (int)(i / Ccols), c = (int)(i % Ccols);
  out[(long)c * R + r] = (bf16_t)in[i];
}

// ---------------- bf16 WMMA GEMM: C[MxN] = A[MxK] * Bt[NxK]^T + bias ----------------
// A row-major bf16 (M x K); Bt is B pre-transposed, row-major bf16 (N x K); C fp32.
// Block: 32x8 = 8 waves. Each wave owns a 32x16 output tile (2 WMMA accs) ->
// block tile is 256 x 16. K is a template constant so the K-loop fully unrolls.
#define BK 32
#define BN 16
#define BMR 256   // rows per block (8 waves * 32 rows)

template<int K>
__global__ __launch_bounds__(256)
void gemm_bf16_wmma_kernel(const bf16_t* __restrict__ A, const bf16_t* __restrict__ Bt,
                           const float* __restrict__ bias, float* __restrict__ C,
                           int M, int Nn)
{
  __shared__ __align__(16) bf16_t sA[BMR][BK + 8];
  __shared__ __align__(16) bf16_t sB[BN][BK + 8];

  const int lane = threadIdx.x;          // 0..31
  const int wave = threadIdx.y;          // 0..7
  const int tid  = wave * 32 + lane;
  const int m0 = blockIdx.y * BMR;
  const int n0 = blockIdx.x * BN;

  v8f acc0 = {}, acc1 = {};
#pragma unroll
  for (int k0 = 0; k0 < K; k0 += BK) {
    // stage A tile: 256 rows x 32 k as 1024 x b128 chunks.
    // Out-of-range rows are CLAMPED (uniform EXEC, no branches); their results
    // are discarded by the guarded store at the end.
#ifdef HAVE_ASYNC_LDS
#pragma unroll
    for (int t = 0; t < 4; ++t) {
      int i = tid + t * 256;
      int r = i >> 2, kc = (i & 3) * 8;
      int gr = m0 + r; gr = gr < M ? gr : (M - 1);
      async_copy_b128(&A[(long)gr * K + (k0 + kc)], &sA[r][kc]);
    }
    if (tid < 64) {
      int nn = tid >> 2, kc = (tid & 3) * 8;
      async_copy_b128(&Bt[(long)(n0 + nn) * K + (k0 + kc)], &sB[nn][kc]);
    }
    __builtin_amdgcn_s_wait_asynccnt(0);
#else
#pragma unroll
    for (int t = 0; t < 4; ++t) {
      int i = tid + t * 256;
      int r = i >> 2, kc = (i & 3) * 8;
      int gr = m0 + r; gr = gr < M ? gr : (M - 1);
      *(v8bf*)&sA[r][kc] = *(const v8bf*)&A[(long)gr * K + (k0 + kc)];
    }
    if (tid < 64) {
      int nn = tid >> 2, kc = (tid & 3) * 8;
      *(v8bf*)&sB[nn][kc] = *(const v8bf*)&Bt[(long)(n0 + nn) * K + (k0 + kc)];
    }
#endif
    if (k0 + BK < K) {
      int pr = m0 + (tid & (BMR - 1)); if (pr >= M) pr = M - 1;
      __builtin_prefetch(&A[(long)pr * K + (k0 + BK)], 0, 3);   // near-temporal
    }
    __syncthreads();

    // Fragments per CDNA5 16-bit WMMA VGPR layout:
    //  A 16x32: lanes 0-15 -> row=lane, K {0..7,16..23}; lanes 16-31 -> K {8..15,24..31}
    //  B 32x16: lanes 0-15 -> col=lane, K {0..7,16..23}; lanes 16-31 -> K {8..15,24..31}
    union { v16bf v; v8bf h[2]; } fa0, fa1, fb;
    {
      int rr = wave * 32 + (lane & 15);
      int kb = (lane >> 4) * 8;
      fa0.h[0] = *(const v8bf*)&sA[rr][kb];
      fa0.h[1] = *(const v8bf*)&sA[rr][kb + 16];
      fa1.h[0] = *(const v8bf*)&sA[rr + 16][kb];
      fa1.h[1] = *(const v8bf*)&sA[rr + 16][kb + 16];
      int col = lane & 15;
      int kc2 = (lane >> 4) * 8;
      fb.h[0] = *(const v8bf*)&sB[col][kc2];
      fb.h[1] = *(const v8bf*)&sB[col][kc2 + 16];
    }
    acc0 = __builtin_amdgcn_wmma_f32_16x16x32_bf16(false, fa0.v, false, fb.v,
                                                   (short)0, acc0, false, false);
    acc1 = __builtin_amdgcn_wmma_f32_16x16x32_bf16(false, fa1.v, false, fb.v,
                                                   (short)0, acc1, false, false);
    __syncthreads();
  }

  // C/D layout: lanes 0-15 -> N=lane, M=v; lanes 16-31 -> N=lane-16, M=8+v
  int col = lane & 15;
  int mb  = (lane >> 4) * 8;
  float bv = bias[n0 + col];
#pragma unroll
  for (int v = 0; v < 8; ++v) {
    int gm0 = m0 + wave * 32 + mb + v;
    if (gm0 < M) C[(long)gm0 * Nn + (n0 + col)] = acc0[v] + bv;
    int gm1 = gm0 + 16;
    if (gm1 < M) C[(long)gm1 * Nn + (n0 + col)] = acc1[v] + bv;
  }
}

// ---------------- GATv2 edge kernels ----------------
// One wave (32 lanes) per edge. Templated on heads/ch so channel loops unroll
// and all gathers are float4 (b128).
template<int Hh, int Cc>
__global__ __launch_bounds__(256)
void gat_logits_kernel(const float* __restrict__ xl, const float* __restrict__ xr,
                       const float* __restrict__ We, const float* __restrict__ att,
                       const float* __restrict__ ea, const int* __restrict__ src,
                       const int* __restrict__ dst, float* __restrict__ logits,
                       unsigned* __restrict__ nmax, int E_)
{
  constexpr int F   = Hh * Cc;
  constexpr int CPL = F / 32;       // channels per lane
  constexpr int NV  = CPL / 4;      // float4 chunks per lane
  constexpr int LPH = 32 / Hh;      // lanes per head (wave32)
  int e    = (blockIdx.x * blockDim.x + threadIdx.x) >> 5;
  int lane = threadIdx.x & 31;
  if (e >= E_) return;
  int s = src[e], d = dst[e];
  float a = ea[e];
  int c0 = lane * CPL;
  const float4* xl4  = (const float4*)(xl + (long)s * F + c0);
  const float4* xr4  = (const float4*)(xr + (long)d * F + c0);
  const float4* We4  = (const float4*)(We + c0);
  const float4* att4 = (const float4*)(att + c0);
  float sum = 0.f;
#pragma unroll
  for (int i = 0; i < NV; ++i) {
    float4 va = xl4[i], vb = xr4[i], vw = We4[i], vt = att4[i];
    sum += leaky(va.x + vb.x + a * vw.x) * vt.x;
    sum += leaky(va.y + vb.y + a * vw.y) * vt.y;
    sum += leaky(va.z + vb.z + a * vw.z) * vt.z;
    sum += leaky(va.w + vb.w + a * vw.w) * vt.w;
  }
#pragma unroll
  for (int off = 1; off < LPH; off <<= 1)
    sum += __shfl_xor(sum, off, 32);
  if ((lane & (LPH - 1)) == 0) {
    int h = lane / LPH;
    logits[(long)e * Hh + h] = sum;
    atomicMax(&nmax[(long)d * Hh + h], f2ord(sum));
  }
}

__global__ void gat_expnorm_kernel(float* __restrict__ logits, const unsigned* __restrict__ nmax,
                                   float* __restrict__ den, const int* __restrict__ dst,
                                   long total, int Hh)
{
  long i = (long)blockIdx.x * blockDim.x + threadIdx.x;
  if (i >= total) return;
  long e = i / Hh; int h = (int)(i - e * Hh);
  int d = dst[e];
  float mx = ord2f(nmax[(long)d * Hh + h]);
  float ex = __expf(logits[i] - mx);
  logits[i] = ex;                      // reuse buffer as numerator
  atomicAdd(&den[(long)d * Hh + h], ex);
}

template<int Hh, int Cc>
__global__ __launch_bounds__(256)
void gat_aggregate_kernel(const float* __restrict__ ex, const float* __restrict__ den,
                          const float* __restrict__ xl, const int* __restrict__ src,
                          const int* __restrict__ dst, float* __restrict__ out, int E_)
{
  constexpr int F   = Hh * Cc;
  constexpr int CPL = F / 32;
  constexpr int NV  = CPL / 4;
  int e    = (blockIdx.x * blockDim.x + threadIdx.x) >> 5;
  int lane = threadIdx.x & 31;
  if (e >= E_) return;
  int s = src[e], d = dst[e];
  int c0 = lane * CPL;
  int h  = c0 / Cc;                    // all CPL channels stay within one head
  float alpha = ex[(long)e * Hh + h] / (den[(long)d * Hh + h] + 1e-16f);
  const float4* xl4 = (const float4*)(xl + (long)s * F + c0);
  float* ob = out + (long)d * F + c0;
#pragma unroll
  for (int i = 0; i < NV; ++i) {
    float4 v = xl4[i];
    atomicAdd(ob + i * 4 + 0, alpha * v.x);
    atomicAdd(ob + i * 4 + 1, alpha * v.y);
    atomicAdd(ob + i * 4 + 2, alpha * v.z);
    atomicAdd(ob + i * 4 + 3, alpha * v.w);
  }
}

// ---------------- BatchNorm (training batch stats) + PReLU ----------------
__global__ void bn_stats_kernel(const float* __restrict__ h, const float* __restrict__ cb,
                                float* __restrict__ sums, float* __restrict__ sqs,
                                int n, int Cfeat, int rowsPerBlock)
{
  int col = blockIdx.x * blockDim.x + threadIdx.x;
  if (col >= Cfeat) return;
  int r0 = blockIdx.y * rowsPerBlock;
  int r1 = r0 + rowsPerBlock; if (r1 > n) r1 = n;
  float cbv = cb[col];
  float s = 0.f, q = 0.f;
  for (int r = r0; r < r1; ++r) {
    float v = h[(long)r * Cfeat + col] + cbv;
    s += v; q += v * v;
  }
  atomicAdd(&sums[col], s);
  atomicAdd(&sqs[col], q);
}

__global__ void bn_prelu_bf16_kernel(const float* __restrict__ h, const float* __restrict__ cb,
                                     const float* __restrict__ sums, const float* __restrict__ sqs,
                                     const float* __restrict__ g, const float* __restrict__ b,
                                     const float* __restrict__ pa, bf16_t* __restrict__ outb,
                                     int n, int Cfeat)
{
  long i = ((long)blockIdx.x * blockDim.x + threadIdx.x) * 4;
  if (i >= (long)n * Cfeat) return;
  int col = (int)(i % Cfeat);
  float4 hv = *(const float4*)(h + i);
  float4 cv = *(const float4*)(cb + col);
  float4 gv = *(const float4*)(g + col);
  float4 bv = *(const float4*)(b + col);
  float4 s4 = *(const float4*)(sums + col);
  float4 q4 = *(const float4*)(sqs + col);
  float a = pa[0];
  float inv = 1.f / (float)n;
  v4bf o;
  {
    float mu = s4.x * inv, var = q4.x * inv - mu * mu;
    float y = (hv.x + cv.x - mu) * rsqrtf(var + BNEPS) * gv.x + bv.x;
    o.x = (bf16_t)(y > 0.f ? y : a * y);
  }
  {
    float mu = s4.y * inv, var = q4.y * inv - mu * mu;
    float y = (hv.y + cv.y - mu) * rsqrtf(var + BNEPS) * gv.y + bv.y;
    o.y = (bf16_t)(y > 0.f ? y : a * y);
  }
  {
    float mu = s4.z * inv, var = q4.z * inv - mu * mu;
    float y = (hv.z + cv.z - mu) * rsqrtf(var + BNEPS) * gv.z + bv.z;
    o.z = (bf16_t)(y > 0.f ? y : a * y);
  }
  {
    float mu = s4.w * inv, var = q4.w * inv - mu * mu;
    float y = (hv.w + cv.w - mu) * rsqrtf(var + BNEPS) * gv.w + bv.w;
    o.w = (bf16_t)(y > 0.f ? y : a * y);
  }
  *(v4bf*)(outb + i) = o;
}

__global__ void bn_prelu_add_kernel(const float* __restrict__ h, const float* __restrict__ cb,
                                    const float* __restrict__ sums, const float* __restrict__ sqs,
                                    const float* __restrict__ g, const float* __restrict__ b,
                                    const float* __restrict__ pa, const float* __restrict__ skip,
                                    float* __restrict__ out, int n, int Cfeat)
{
  long i = ((long)blockIdx.x * blockDim.x + threadIdx.x) * 4;
  if (i >= (long)n * Cfeat) return;
  int col = (int)(i % Cfeat);
  float4 hv = *(const float4*)(h + i);
  float4 cv = *(const float4*)(cb + col);
  float4 gv = *(const float4*)(g + col);
  float4 bv = *(const float4*)(b + col);
  float4 s4 = *(const float4*)(sums + col);
  float4 q4 = *(const float4*)(sqs + col);
  float4 sk = *(const float4*)(skip + i);
  float a = pa[0];
  float inv = 1.f / (float)n;
  float4 r;
  {
    float mu = s4.x * inv, var = q4.x * inv - mu * mu;
    float y = (hv.x + cv.x - mu) * rsqrtf(var + BNEPS) * gv.x + bv.x;
    r.x = (y > 0.f ? y : a * y) + sk.x;
  }
  {
    float mu = s4.y * inv, var = q4.y * inv - mu * mu;
    float y = (hv.y + cv.y - mu) * rsqrtf(var + BNEPS) * gv.y + bv.y;
    r.y = (y > 0.f ? y : a * y) + sk.y;
  }
  {
    float mu = s4.z * inv, var = q4.z * inv - mu * mu;
    float y = (hv.z + cv.z - mu) * rsqrtf(var + BNEPS) * gv.z + bv.z;
    r.z = (y > 0.f ? y : a * y) + sk.z;
  }
  {
    float mu = s4.w * inv, var = q4.w * inv - mu * mu;
    float y = (hv.w + cv.w - mu) * rsqrtf(var + BNEPS) * gv.w + bv.w;
    r.w = (y > 0.f ? y : a * y) + sk.w;
  }
  *(float4*)(out + i) = r;
}

// ---------------- launcher ----------------
extern "C" void kernel_launch(void* const* d_in, const int* in_sizes, int n_in,
                              void* d_out, int out_size, void* d_ws, size_t ws_size,
                              hipStream_t stream)
{
  (void)in_sizes; (void)n_in; (void)out_size; (void)ws_size;
  const float* x     = (const float*)d_in[0];
  const int*   ei    = (const int*)  d_in[1];
  const float* ea    = (const float*)d_in[2];
  const float* Wl1   = (const float*)d_in[3];
  const float* bl1   = (const float*)d_in[4];
  const float* Wr1   = (const float*)d_in[5];
  const float* br1   = (const float*)d_in[6];
  const float* We1   = (const float*)d_in[7];
  const float* att1  = (const float*)d_in[8];
  const float* cb1   = (const float*)d_in[9];
  const float* bng1  = (const float*)d_in[10];
  const float* bnb1  = (const float*)d_in[11];
  const float* pa    = (const float*)d_in[12];
  const float* Wl2   = (const float*)d_in[13];
  const float* bl2   = (const float*)d_in[14];
  const float* Wr2   = (const float*)d_in[15];
  const float* br2   = (const float*)d_in[16];
  const float* We2   = (const float*)d_in[17];
  const float* att2  = (const float*)d_in[18];
  const float* cb2   = (const float*)d_in[19];
  const float* bng2  = (const float*)d_in[20];
  const float* bnb2  = (const float*)d_in[21];
  const float* skipW = (const float*)d_in[22];
  const float* skipb = (const float*)d_in[23];

  const int NN = 50000, EE = 800000, INF = 128, F1 = 256, F2 = 128;
  const int H1 = 4, H2 = 1;
  const int* src = ei;
  const int* dst = ei + EE;

  size_t off = 0;
  auto alloc = [&](size_t bytes) -> void* {
    void* p = (char*)d_ws + off;
    off += (bytes + 255) & ~(size_t)255;
    return p;
  };
  bf16_t* xb     = (bf16_t*)alloc((size_t)NN * INF * 2);
  bf16_t* Wl1t   = (bf16_t*)alloc((size_t)INF * F1 * 2);   // F1 x INF (transposed)
  bf16_t* Wr1t   = (bf16_t*)alloc((size_t)INF * F1 * 2);
  bf16_t* Wl2t   = (bf16_t*)alloc((size_t)F1 * F2 * 2);    // F2 x F1
  bf16_t* Wr2t   = (bf16_t*)alloc((size_t)F1 * F2 * 2);
  bf16_t* skipWt = (bf16_t*)alloc((size_t)INF * F2 * 2);   // F2 x INF
  float* xl1     = (float*)alloc((size_t)NN * F1 * 4);
  float* xr1     = (float*)alloc((size_t)NN * F1 * 4);
  float* logits1 = (float*)alloc((size_t)EE * H1 * 4);
  unsigned* nmax1= (unsigned*)alloc((size_t)NN * H1 * 4);
  float* den1    = (float*)alloc((size_t)NN * H1 * 4);
  float* out1    = (float*)alloc((size_t)NN * F1 * 4);
  float* sums1   = (float*)alloc((size_t)F1 * 4);
  float* sqs1    = (float*)alloc((size_t)F1 * 4);
  bf16_t* h1b    = (bf16_t*)alloc((size_t)NN * F1 * 2);
  float* xl2     = (float*)alloc((size_t)NN * F2 * 4);
  float* xr2     = (float*)alloc((size_t)NN * F2 * 4);
  float* skipO   = (float*)alloc((size_t)NN * F2 * 4);
  float* logits2 = (float*)alloc((size_t)EE * H2 * 4);
  unsigned* nmax2= (unsigned*)alloc((size_t)NN * H2 * 4);
  float* den2    = (float*)alloc((size_t)NN * H2 * 4);
  float* out2    = (float*)alloc((size_t)NN * F2 * 4);
  float* sums2   = (float*)alloc((size_t)F2 * 4);
  float* sqs2    = (float*)alloc((size_t)F2 * 4);

  auto cvt = [&](const float* in, bf16_t* outp, long n) {
    f32_to_bf16_kernel<<<dim3((unsigned)((n / 4 + 255) / 256)), dim3(256), 0, stream>>>(in, outp, n);
  };
  auto cvtT = [&](const float* in, bf16_t* outp, int R, int Ccols) {
    long n = (long)R * Ccols;
    f32_to_bf16_T_kernel<<<dim3((unsigned)((n + 255) / 256)), dim3(256), 0, stream>>>(in, outp, R, Ccols);
  };
  auto fill = [&](void* p, unsigned v, long n) {
    fill_u32_kernel<<<dim3((unsigned)((n + 255) / 256)), dim3(256), 0, stream>>>((unsigned*)p, v, n);
  };

  const unsigned ORD_NEG_INF = 0x007FFFFFu;   // f2ord(-inf)

  // ---- phase 0: converts (+ weight transpose) + inits ----
  cvt(x, xb, (long)NN * INF);
  cvtT(Wl1, Wl1t, INF, F1);
  cvtT(Wr1, Wr1t, INF, F1);
  cvtT(Wl2, Wl2t, F1, F2);
  cvtT(Wr2, Wr2t, F1, F2);
  cvtT(skipW, skipWt, INF, F2);
  fill(out1, 0u, (long)NN * F1);
  fill(den1, 0u, (long)NN * H1);
  fill(nmax1, ORD_NEG_INF, (long)NN * H1);
  fill(sums1, 0u, F1); fill(sqs1, 0u, F1);
  fill(out2, 0u, (long)NN * F2);
  fill(den2, 0u, (long)NN * H2);
  fill(nmax2, ORD_NEG_INF, (long)NN * H2);
  fill(sums2, 0u, F2); fill(sqs2, 0u, F2);

  // ---- layer 1 GEMMs (K = 128) ----
  {
    dim3 blk(32, 8);
    dim3 grd((unsigned)(F1 / 16), (unsigned)((NN + BMR - 1) / BMR));
    gemm_bf16_wmma_kernel<128><<<grd, blk, 0, stream>>>(xb, Wl1t, bl1, xl1, NN, F1);
    gemm_bf16_wmma_kernel<128><<<grd, blk, 0, stream>>>(xb, Wr1t, br1, xr1, NN, F1);
  }

  // ---- layer 1 attention ----
  {
    dim3 blk(256); dim3 grd((unsigned)(EE / 8));          // 8 waves/block, 1 edge/wave
    gat_logits_kernel<4, 64><<<grd, blk, 0, stream>>>(xl1, xr1, We1, att1, ea, src, dst,
                                                      logits1, nmax1, EE);
    long tot = (long)EE * H1;
    gat_expnorm_kernel<<<dim3((unsigned)((tot + 255) / 256)), blk, 0, stream>>>(
        logits1, nmax1, den1, dst, tot, H1);
    gat_aggregate_kernel<4, 64><<<grd, blk, 0, stream>>>(logits1, den1, xl1, src, dst,
                                                         out1, EE);
  }

  // ---- BN1 + PReLU -> bf16 h1 ----
  {
    int rpb = (NN + 63) / 64;
    bn_stats_kernel<<<dim3(F1 / 128, 64), dim3(128), 0, stream>>>(out1, cb1, sums1, sqs1, NN, F1, rpb);
    long tot = (long)NN * F1;
    bn_prelu_bf16_kernel<<<dim3((unsigned)((tot / 4 + 255) / 256)), dim3(256), 0, stream>>>(
        out1, cb1, sums1, sqs1, bng1, bnb1, pa, h1b, NN, F1);
  }

  // ---- layer 2 GEMMs (K = 256) + skip GEMM (K = 128) ----
  {
    dim3 blk(32, 8);
    dim3 grd((unsigned)(F2 / 16), (unsigned)((NN + BMR - 1) / BMR));
    gemm_bf16_wmma_kernel<256><<<grd, blk, 0, stream>>>(h1b, Wl2t, bl2, xl2, NN, F2);
    gemm_bf16_wmma_kernel<256><<<grd, blk, 0, stream>>>(h1b, Wr2t, br2, xr2, NN, F2);
    gemm_bf16_wmma_kernel<128><<<grd, blk, 0, stream>>>(xb, skipWt, skipb, skipO, NN, F2);
  }

  // ---- layer 2 attention ----
  {
    dim3 blk(256); dim3 grd((unsigned)(EE / 8));
    gat_logits_kernel<1, 128><<<grd, blk, 0, stream>>>(xl2, xr2, We2, att2, ea, src, dst,
                                                       logits2, nmax2, EE);
    long tot = (long)EE * H2;
    gat_expnorm_kernel<<<dim3((unsigned)((tot + 255) / 256)), blk, 0, stream>>>(
        logits2, nmax2, den2, dst, tot, H2);
    gat_aggregate_kernel<1, 128><<<grd, blk, 0, stream>>>(logits2, den2, xl2, src, dst,
                                                          out2, EE);
  }

  // ---- BN2 + PReLU + skip-add -> d_out ----
  {
    int rpb = (NN + 63) / 64;
    bn_stats_kernel<<<dim3(F2 / 128, 64), dim3(128), 0, stream>>>(out2, cb2, sums2, sqs2, NN, F2, rpb);
    long tot = (long)NN * F2;
    bn_prelu_add_kernel<<<dim3((unsigned)((tot / 4 + 255) / 256)), dim3(256), 0, stream>>>(
        out2, cb2, sums2, sqs2, bng2, bnb2, pa, skipO, (float*)d_out, NN, F2);
  }
}
